// GRUScratch_44856638439609
// MI455X (gfx1250) — compile-verified, matrix operand order checked by reading, not run on hardware
//
#include <hip/hip_runtime.h>
#include <hip/hip_bf16.h>

// GRU: T=1024 steps, B=64, D=1024.
// Phase 1: Xz/Xr/Xh = inputs @ W_x* + b  (bf16-WMMA GEMM, software-pipelined
//          b128 fragment loads so WMMAs overlap with next k-step's loads).
// Phase 2: 1024 graph-replayed step kernels. Each output 16x16 tile is split
//          across 2 waves in K (48 WMMAs each instead of 96 -> half the serial
//          chain in the latency-bound scan), partial sums reduced through LDS,
//          then fused sigmoid/tanh gate math. H_t is written as f32 (d_out)
//          and bf16 (A operand of step t+1).

#define T_STEPS 1024
#define BATCH   64
#define DIM     1024
#define KBLKS   (DIM / 32)   // 32 k-steps of 32 per WMMA
#define NTILES  (DIM / 16)   // 64 column tiles
#define KHALF   (KBLKS / 2)  // 16 k-steps per wave in the K-split step kernel

typedef __attribute__((ext_vector_type(16))) __bf16 v16bf;
typedef __attribute__((ext_vector_type(8)))  float  v8f;

union ABFrag {
    v16bf v;
    unsigned int u[8];
    uint4 q[2];
};

__device__ __forceinline__ unsigned int pk_bf16(float lo, float hi) {
    union { float f; unsigned int u; } a, b;
    a.f = lo; b.f = hi;
    unsigned int ul = (a.u + 0x7FFFu + ((a.u >> 16) & 1u)) >> 16;   // RNE truncation
    unsigned int uh = (b.u + 0x7FFFu + ((b.u >> 16) & 1u)) >> 16;
    return (ul & 0xFFFFu) | (uh << 16);
}

__device__ __forceinline__ unsigned short f32_to_bf16(float f) {
    union { float f; unsigned int u; } a;
    a.f = f;
    return (unsigned short)((a.u + 0x7FFFu + ((a.u >> 16) & 1u)) >> 16);
}

// A fragment from a row-major bf16 matrix (ld = DIM). ISA 16-bit A layout:
// lanes 0-15 -> M=lane, K {0..7,16..23}; lanes 16-31 -> M=lane-16, K {8..15,24..31}.
__device__ __forceinline__ v16bf load_a_bf16(const unsigned short* __restrict__ A,
                                             int m0, int kbase, int lane) {
    int l16 = lane & 15;
    int hi  = (lane >> 4) & 1;
    const unsigned short* p = A + (size_t)(m0 + l16) * DIM + kbase + hi * 8;
    ABFrag r;
    r.q[0] = *(const uint4*)(p);        // K local 0..7   (+8 if hi)
    r.q[1] = *(const uint4*)(p + 16);   // K local 16..23 (+8 if hi)
    return r.v;
}

// B fragments pre-packed per (ntile, kblk): 32 lanes x 8 dwords, contiguous.
__device__ __forceinline__ v16bf load_b_packed(const unsigned int* __restrict__ P,
                                               int ntile, int kblk, int lane) {
    const uint4* q = (const uint4*)(P + ((((size_t)ntile * KBLKS) + kblk) * 32 + lane) * 8);
    ABFrag r;
    r.q[0] = q[0];
    r.q[1] = q[1];
    return r.v;
}

__device__ __forceinline__ float sigmoidf(float x) {
    return 1.0f / (1.0f + __expf(-x));
}

#define WMMA_BF16(C, A, B) \
    __builtin_amdgcn_wmma_f32_16x16x32_bf16(false, (A), false, (B), (short)0, (C), false, false)

// ---------------------------------------------------------------------------
// Pack a row-major f32 DxD weight matrix into bf16 WMMA B-fragment layout.
// B layout mirrors A: lanes 0-15 -> N=lane, K {0..7,16..23}; lanes 16-31 ->
// N=lane-16, K {8..15,24..31}; VGPR v holds a K pair (even K in [15:0]).
// ---------------------------------------------------------------------------
__global__ void gru_pack_w(const float* __restrict__ w0, const float* __restrict__ w1,
                           const float* __restrict__ w2, const float* __restrict__ w3,
                           const float* __restrict__ w4, const float* __restrict__ w5,
                           unsigned int* __restrict__ dst) {
    int wave = (blockIdx.x * blockDim.x + threadIdx.x) >> 5;
    int lane = threadIdx.x & 31;
    if (wave >= 6 * NTILES * KBLKS) return;
    int mat   = wave / (NTILES * KBLKS);
    int rem   = wave % (NTILES * KBLKS);
    int ntile = rem / KBLKS;
    int kblk  = rem % KBLKS;
    const float* W = mat == 0 ? w0 : mat == 1 ? w1 : mat == 2 ? w2
                   : mat == 3 ? w3 : mat == 4 ? w4 : w5;
    unsigned int* out = dst + (size_t)mat * ((size_t)DIM * DIM / 2)
                      + ((((size_t)ntile * KBLKS) + kblk) * 32 + lane) * 8;
    int n  = ntile * 16 + (lane & 15);
    int hi = (lane >> 4) & 1;
#pragma unroll
    for (int v = 0; v < 8; ++v) {
        int kloc = (v < 4 ? 2 * v : 16 + 2 * (v - 4)) + hi * 8;
        int k = kblk * 32 + kloc;
        out[v] = pk_bf16(W[(size_t)k * DIM + n], W[(size_t)(k + 1) * DIM + n]);
    }
}

// Bulk f32 -> bf16 row-major conversion (8 elements per thread, b128 in/out).
__global__ void gru_cvt_bf16(const float* __restrict__ src,
                             unsigned short* __restrict__ dst, long long n) {
    long long i = ((long long)blockIdx.x * blockDim.x + threadIdx.x) * 8;
    if (i >= n) return;
    float4 f0 = *(const float4*)(src + i);
    float4 f1 = *(const float4*)(src + i + 4);
    uint4 o;
    o.x = pk_bf16(f0.x, f0.y);
    o.y = pk_bf16(f0.z, f0.w);
    o.z = pk_bf16(f1.x, f1.y);
    o.w = pk_bf16(f1.z, f1.w);
    *(uint4*)(dst + i) = o;
}

// ---------------------------------------------------------------------------
// Phase 1: Xg = inputs_bf16 @ W_xg + b_g for g in {z,r,h}. One wave per 16x16
// output tile, all 3 gates; A fragment reused 3x per k-step. Fragments for
// k-step i+1 are prefetched before the WMMAs of k-step i (partial waits).
// ---------------------------------------------------------------------------
__global__ void __launch_bounds__(256)
gru_xproj(const unsigned short* __restrict__ Xbf,
          const unsigned int* __restrict__ Pz, const unsigned int* __restrict__ Pr,
          const unsigned int* __restrict__ Ph,
          const float* __restrict__ bz, const float* __restrict__ br,
          const float* __restrict__ bh,
          float* __restrict__ Xz, float* __restrict__ Xr, float* __restrict__ Xh) {
    int wave  = (blockIdx.x * blockDim.x + threadIdx.x) >> 5;
    int lane  = threadIdx.x & 31;
    int ntile = wave & (NTILES - 1);
    int mtile = wave >> 6;              // 0 .. T*B/16-1
    int m0 = mtile * 16, n0 = ntile * 16;

    v8f cz = {}, cr = {}, ch = {};
    v16bf a  = load_a_bf16(Xbf, m0, 0, lane);
    v16bf b0 = load_b_packed(Pz, ntile, 0, lane);
    v16bf b1 = load_b_packed(Pr, ntile, 0, lane);
    v16bf b2 = load_b_packed(Ph, ntile, 0, lane);
#pragma unroll 4
    for (int kb = 1; kb < KBLKS; ++kb) {
        v16bf an  = load_a_bf16(Xbf, m0, kb * 32, lane);
        v16bf b0n = load_b_packed(Pz, ntile, kb, lane);
        v16bf b1n = load_b_packed(Pr, ntile, kb, lane);
        v16bf b2n = load_b_packed(Ph, ntile, kb, lane);
        cz = WMMA_BF16(cz, a, b0);
        cr = WMMA_BF16(cr, a, b1);
        ch = WMMA_BF16(ch, a, b2);
        a = an; b0 = b0n; b1 = b1n; b2 = b2n;
    }
    cz = WMMA_BF16(cz, a, b0);
    cr = WMMA_BF16(cr, a, b1);
    ch = WMMA_BF16(ch, a, b2);

    int l16 = lane & 15, hi = (lane >> 4) & 1;
    int n = n0 + l16;
    float vbz = bz[n], vbr = br[n], vbh = bh[n];
#pragma unroll
    for (int r = 0; r < 8; ++r) {
        // C/D layout: VGPR r, lanes 0-15 -> M=m0+r, lanes 16-31 -> M=m0+8+r
        size_t idx = (size_t)(m0 + r + hi * 8) * DIM + n;
        Xz[idx] = cz[r] + vbz;
        Xr[idx] = cr[r] + vbr;
        Xh[idx] = ch[r] + vbh;
    }
}

// ---------------------------------------------------------------------------
// Phase 2: one fused timestep. 512 waves: each 16x16 output tile is computed
// by a pair of waves splitting K (16 k-steps each), with an LDS reduction.
// Even wave of the pair then applies the gate math and stores H_t (f32+bf16).
// ---------------------------------------------------------------------------
__global__ void __launch_bounds__(256)
gru_step(const unsigned short* __restrict__ Hbf,   // H_{t-1}, bf16 row-major
         const float* __restrict__ Hprev,          // H_{t-1}, f32
         const float* __restrict__ Xz, const float* __restrict__ Xr,
         const float* __restrict__ Xh,
         const unsigned int* __restrict__ Pz, const unsigned int* __restrict__ Pr,
         const unsigned int* __restrict__ Ph,
         float* __restrict__ Hnew,                 // d_out slice t
         unsigned short* __restrict__ HbfOut) {    // bf16 H_t for step t+1
    __shared__ float red[4][24][32];               // 12KB: per-pair partial C

    int wave  = threadIdx.x >> 5;                  // 0..7
    int lane  = threadIdx.x & 31;
    int gwave = blockIdx.x * 8 + wave;             // 0..511
    int pair  = gwave >> 1;                        // tile id: 0..255
    int kh    = gwave & 1;                         // K half
    int mtile = pair & 3;                          // BATCH/16 = 4
    int ntile = pair >> 2;                         // 0..63
    int m0 = mtile * 16, n0 = ntile * 16;
    int kb0 = kh * KHALF;

    v8f cz = {}, cr = {}, ch = {};
    v16bf a  = load_a_bf16(Hbf, m0, kb0 * 32, lane);
    v16bf b0 = load_b_packed(Pz, ntile, kb0, lane);
    v16bf b1 = load_b_packed(Pr, ntile, kb0, lane);
    v16bf b2 = load_b_packed(Ph, ntile, kb0, lane);
#pragma unroll 4
    for (int i = 1; i < KHALF; ++i) {
        v16bf an  = load_a_bf16(Hbf, m0, (kb0 + i) * 32, lane);
        v16bf b0n = load_b_packed(Pz, ntile, kb0 + i, lane);
        v16bf b1n = load_b_packed(Pr, ntile, kb0 + i, lane);
        v16bf b2n = load_b_packed(Ph, ntile, kb0 + i, lane);
        cz = WMMA_BF16(cz, a, b0);
        cr = WMMA_BF16(cr, a, b1);
        ch = WMMA_BF16(ch, a, b2);
        a = an; b0 = b0n; b1 = b1n; b2 = b2n;
    }
    cz = WMMA_BF16(cz, a, b0);
    cr = WMMA_BF16(cr, a, b1);
    ch = WMMA_BF16(ch, a, b2);

    int pib = wave >> 1;                           // pair index in block: 0..3
    if (kh == 1) {
#pragma unroll
        for (int r = 0; r < 8; ++r) {
            red[pib][r][lane]      = cz[r];
            red[pib][8 + r][lane]  = cr[r];
            red[pib][16 + r][lane] = ch[r];
        }
    }
    __syncthreads();
    if (kh == 0) {
#pragma unroll
        for (int r = 0; r < 8; ++r) {
            cz[r] += red[pib][r][lane];
            cr[r] += red[pib][8 + r][lane];
            ch[r] += red[pib][16 + r][lane];
        }
        int l16 = lane & 15, hi = (lane >> 4) & 1;
#pragma unroll
        for (int r = 0; r < 8; ++r) {
            size_t idx = (size_t)(m0 + r + hi * 8) * DIM + n0 + l16;
            float xz = Xz[idx], xr = Xr[idx], xh = Xh[idx];
            float hp = Hprev[idx];
            float z  = sigmoidf(xz + cz[r]);
            float rr = sigmoidf(xr + cr[r]);
            float ht = tanhf(xh + rr * ch[r]);
            float hn = z * hp + (1.0f - z) * ht;
            Hnew[idx]   = hn;
            HbfOut[idx] = f32_to_bf16(hn);
        }
    }
}

__global__ void gru_zero_f32(float* __restrict__ p, int n) {
    int i = blockIdx.x * blockDim.x + threadIdx.x;
    if (i < n) p[i] = 0.0f;
}

__global__ void gru_zero_bf16(unsigned short* __restrict__ p, int n) {
    int i = blockIdx.x * blockDim.x + threadIdx.x;
    if (i < n) p[i] = 0;
}

__global__ void gru_copy(const float* __restrict__ src, float* __restrict__ dst, int n) {
    int i = blockIdx.x * blockDim.x + threadIdx.x;
    if (i < n) dst[i] = src[i];
}

extern "C" void kernel_launch(void* const* d_in, const int* in_sizes, int n_in,
                              void* d_out, int out_size, void* d_ws, size_t ws_size,
                              hipStream_t stream) {
    const float* inputs = (const float*)d_in[0];
    const float* W_xz   = (const float*)d_in[1];
    const float* W_hz   = (const float*)d_in[2];
    const float* b_z    = (const float*)d_in[3];
    const float* W_xr   = (const float*)d_in[4];
    const float* W_hr   = (const float*)d_in[5];
    const float* b_r    = (const float*)d_in[6];
    const float* W_xh   = (const float*)d_in[7];
    const float* W_hh   = (const float*)d_in[8];
    const float* b_h    = (const float*)d_in[9];
    float* out = (float*)d_out;   // [T,B,D] outputs then [B,D] H_last

    // Workspace layout:
    //  packed bf16 weights (6 x 2MB), H0 f32 (256KB), H0/H ping-pong bf16
    //  (3 x 128KB), bf16 inputs (128MB), then Xz/Xr/Xh f32 (3 x 268MB).
    char* ws = (char*)d_ws;
    const size_t matU32  = (size_t)DIM * DIM / 2;
    const size_t slice   = (size_t)BATCH * DIM;
    const size_t sliceT  = (size_t)T_STEPS * BATCH * DIM;

    unsigned int* packed = (unsigned int*)ws;
    size_t off = 6ull * matU32 * sizeof(unsigned int);
    float* Hzero = (float*)(ws + off);                     off += slice * sizeof(float);
    unsigned short* HbfZero = (unsigned short*)(ws + off); off += slice * sizeof(unsigned short);
    unsigned short* Hbf0    = (unsigned short*)(ws + off); off += slice * sizeof(unsigned short);
    unsigned short* Hbf1    = (unsigned short*)(ws + off); off += slice * sizeof(unsigned short);
    off = (off + 255) & ~(size_t)255;
    unsigned short* inBf = (unsigned short*)(ws + off);    off += sliceT * sizeof(unsigned short);
    off = (off + 255) & ~(size_t)255;
    float* Xz = (float*)(ws + off);
    float* Xr = Xz + sliceT;
    float* Xh = Xr + sliceT;

    const unsigned int* Pxz = packed + 0 * matU32;
    const unsigned int* Pxr = packed + 1 * matU32;
    const unsigned int* Pxh = packed + 2 * matU32;
    const unsigned int* Phz = packed + 3 * matU32;
    const unsigned int* Phr = packed + 4 * matU32;
    const unsigned int* Phh = packed + 5 * matU32;

    // 1) pack weights into WMMA B-fragment bf16 layout
    {
        int jobs = 6 * NTILES * KBLKS;           // 12288 waves
        gru_pack_w<<<jobs / 8, 256, 0, stream>>>(W_xz, W_xr, W_xh, W_hz, W_hr, W_hh, packed);
    }
    // 2) zero H0 (f32 and bf16)
    gru_zero_f32<<<(int)(slice / 256), 256, 0, stream>>>(Hzero, (int)slice);
    gru_zero_bf16<<<(int)(slice / 256), 256, 0, stream>>>(HbfZero, (int)slice);

    // 3) inputs -> bf16 (one bulk pass; keeps cvt VALU out of the GEMM loops)
    {
        long long n = (long long)sliceT;
        int threads = (int)(n / 8);
        gru_cvt_bf16<<<threads / 256, 256, 0, stream>>>(inputs, inBf, n);
    }

    // 4) phase 1: input projections (compute-bound bf16 WMMA GEMM)
    {
        int waves = (T_STEPS * BATCH / 16) * NTILES;   // 262144 waves
        gru_xproj<<<waves / 8, 256, 0, stream>>>(inBf, Pxz, Pxr, Pxh,
                                                 b_z, b_r, b_h, Xz, Xr, Xh);
    }

    // 5) sequential scan: one fused step kernel per t (graph-replayed).
    //    64 blocks x 8 waves = 512 waves (2 per output tile, K-split).
    for (int t = 0; t < T_STEPS; ++t) {
        const unsigned short* hbfIn = (t == 0) ? HbfZero : ((t & 1) ? Hbf0 : Hbf1);
        unsigned short*       hbfOut = (t & 1) ? Hbf1 : Hbf0;
        const float* hprev = (t == 0) ? Hzero : out + (size_t)(t - 1) * slice;
        gru_step<<<64, 256, 0, stream>>>(hbfIn, hprev,
                                         Xz + (size_t)t * slice,
                                         Xr + (size_t)t * slice,
                                         Xh + (size_t)t * slice,
                                         Phz, Phr, Phh,
                                         out + (size_t)t * slice,
                                         hbfOut);
    }

    // 6) H_last = outputs[T-1]
    gru_copy<<<(int)(slice / 256), 256, 0, stream>>>(out + (size_t)(T_STEPS - 1) * slice,
                                                     out + sliceT, (int)slice);
}
